// MOELayer_33578054320709
// MI455X (gfx1250) — matrix operand order repaired and checked by reading, not compiled, not run on hardware
//
#include <hip/hip_runtime.h>

// ---------------------------------------------------------------------------
// MoE Top-1 layer for MI455X (gfx1250, wave32, WMMA bf16 -> f32 accumulate,
// async global->LDS staging for the bf16 operand)
// B=2, T=2048, M=2048, H=8192, E=8, S=4096, C=512
// ---------------------------------------------------------------------------

#define MOE_E 8
#define MOE_S 4096
#define MOE_M 2048
#define MOE_H 8192
#define MOE_C 512

typedef __attribute__((ext_vector_type(16))) __bf16         v16bf;
typedef __attribute__((ext_vector_type(8)))  float          v8f;
typedef __attribute__((ext_vector_type(16))) unsigned short v16u16;
typedef __attribute__((ext_vector_type(8)))  unsigned short v8u16;

union V16 { v16u16 u; v16bf b; };

__device__ __forceinline__ unsigned short f2bf(float f) {
    // round-to-nearest-even float -> bf16 (bit pattern)
    unsigned u = __float_as_uint(f);
    unsigned r = u + 0x7FFFu + ((u >> 16) & 1u);
    return (unsigned short)(r >> 16);
}

__device__ __forceinline__ unsigned f2bf_pk(float lo, float hi) {
    return (unsigned)f2bf(lo) | ((unsigned)f2bf(hi) << 16);
}

// raw 32-bit LDS byte offset for a __shared__ address (for async-LDS asm)
__device__ __forceinline__ unsigned lds_offset(const void* p) {
    return (unsigned)(size_t)(__attribute__((address_space(3))) const void*)p;
}

// async DMA: 16 bytes global -> LDS, tracked by ASYNCcnt (no VGPR round trip)
__device__ __forceinline__ void async_copy_b128(unsigned lds_off,
                                                const void* gaddr) {
    asm volatile("global_load_async_to_lds_b128 %0, %1, off"
                 :
                 : "v"(lds_off), "v"(gaddr)
                 : "memory");
}

__device__ __forceinline__ void wait_asynccnt0() {
    asm volatile("s_wait_asynccnt 0x0" ::: "memory");
}

// ---------------------------------------------------------------------------
// 0) zero the me[] accumulator
// ---------------------------------------------------------------------------
__global__ void init_me_kernel(float* me) {
    if (threadIdx.x < MOE_E) me[threadIdx.x] = 0.0f;
}

// ---------------------------------------------------------------------------
// 1) gating: logits, softmax, argmax, gates1, me accumulation
// ---------------------------------------------------------------------------
__global__ void __launch_bounds__(256)
gate_kernel(const float* __restrict__ xs, const float* __restrict__ wg,
            int* __restrict__ idx, float* __restrict__ gates1,
            float* __restrict__ me) {
    const int s   = blockIdx.x;
    const int tid = threadIdx.x;
    const float* xrow = xs + (size_t)s * MOE_M;

    float acc[MOE_E];
#pragma unroll
    for (int e = 0; e < MOE_E; ++e) acc[e] = 0.0f;

    for (int m = tid; m < MOE_M; m += 256) {
        float xv = xrow[m];
#pragma unroll
        for (int e = 0; e < MOE_E; ++e) acc[e] += xv * wg[e * MOE_M + m];
    }

    __shared__ float red[256];
    __shared__ float logit[MOE_E];
    for (int e = 0; e < MOE_E; ++e) {
        red[tid] = acc[e];
        __syncthreads();
        for (int off = 128; off > 0; off >>= 1) {
            if (tid < off) red[tid] += red[tid + off];
            __syncthreads();
        }
        if (tid == 0) logit[e] = red[0];
        __syncthreads();
    }

    if (tid == 0) {
        float mx = logit[0];
        int best = 0;
#pragma unroll
        for (int e = 1; e < MOE_E; ++e)
            if (logit[e] > mx) { mx = logit[e]; best = e; }
        float sum = 0.0f;
        float p[MOE_E];
#pragma unroll
        for (int e = 0; e < MOE_E; ++e) { p[e] = expf(logit[e] - mx); sum += p[e]; }
        float inv = 1.0f / sum;
        idx[s]    = best;
        gates1[s] = p[best] * inv;
#pragma unroll
        for (int e = 0; e < MOE_E; ++e) atomicAdd(&me[e], p[e] * inv);
    }
}

// ---------------------------------------------------------------------------
// 2) per-expert cumulative positions -> slot[], inverse map inv[], l_aux
// ---------------------------------------------------------------------------
__global__ void __launch_bounds__(256)
scan_kernel(const int* __restrict__ idx, const float* __restrict__ me,
            int* __restrict__ slot, int* __restrict__ inv,
            float* __restrict__ laux) {
    __shared__ int   sidx[MOE_S];
    __shared__ float sce[MOE_E];
    const int tid = threadIdx.x;

    for (int i = tid; i < MOE_S; i += 256) sidx[i] = idx[i];
    for (int i = tid; i < MOE_E * MOE_C; i += 256) inv[i] = -1;
    __syncthreads();

    if (tid < MOE_E) {
        int cnt = 0;
        for (int s = 0; s < MOE_S; ++s) {
            if (sidx[s] == tid) {
                if (cnt < MOE_C) {
                    slot[s] = tid * MOE_C + cnt;
                    inv[tid * MOE_C + cnt] = s;
                } else {
                    slot[s] = -1;  // capacity overflow -> dropped
                }
                cnt++;
            }
        }
        sce[tid] = (float)cnt;
    }
    __syncthreads();

    if (tid == 0) {
        float a = 0.0f;
#pragma unroll
        for (int e = 0; e < MOE_E; ++e) a += me[e] * sce[e];
        *laux = a * ((float)MOE_E / ((float)MOE_S * (float)MOE_S));
    }
}

// ---------------------------------------------------------------------------
// 3) dispatch: scatter tokens (fp32 -> bf16) into [E*C, M]; zero empty slots
// ---------------------------------------------------------------------------
__global__ void __launch_bounds__(256)
dispatch_kernel(const float* __restrict__ xs, const int* __restrict__ inv,
                unsigned short* __restrict__ disp) {
    const int b = blockIdx.x;           // slot id in [0, E*C)
    const int t = inv[b];
    unsigned* drow = (unsigned*)(disp + (size_t)b * MOE_M);
    if (t >= 0) {
        const float* xrow = xs + (size_t)t * MOE_M;
        for (int m = threadIdx.x; m < MOE_M / 2; m += 256)
            drow[m] = f2bf_pk(xrow[2 * m], xrow[2 * m + 1]);
    } else {
        for (int m = threadIdx.x; m < MOE_M / 2; m += 256) drow[m] = 0u;
    }
}

// ---------------------------------------------------------------------------
// 4) grouped GEMM: D[e] = act(A[e] @ B[e])   (bf16 x bf16 -> f32 accumulate)
//    A: bf16 [Mrows, K] per expert -> staged with global_load_async_to_lds_b128
//    B: f32  [K, N]     per expert -> batched loads, f32->bf16, packed stores
//    128x128 block tile, BK=32, double-buffered LDS, one barrier per K-step,
//    8 waves (2x4), each wave 4x2 tiles of v_wmma_f32_16x16x32_bf16.
// ---------------------------------------------------------------------------
template <bool RELU, bool OUTBF>
__global__ void __launch_bounds__(256)
moe_gemm(const unsigned short* __restrict__ Aall, const float* __restrict__ Ball,
         void* __restrict__ Dall, int Mrows, int N, int K) {
    const int e    = blockIdx.z;
    const int row0 = blockIdx.y * 128;
    const int col0 = blockIdx.x * 128;

    const unsigned short* A  = Aall + (size_t)e * Mrows * K;
    const float*          Bg = Ball + (size_t)e * K * N;

    // double-buffered tiles: A [row][k], B [col][k] (transposed), bf16
    __shared__ __align__(64) unsigned short As[2][128 * 32];
    __shared__ __align__(64) unsigned short Bs[2][128 * 32];

    const int tid  = threadIdx.x;
    const int lane = tid & 31;
    const int wave = tid >> 5;
    const int wm   = wave & 1;   // row half:   wm*64
    const int wn   = wave >> 1;  // col group:  wn*32

    const int mlo = lane & 15;   // row / col within 16
    const int hi  = lane >> 4;   // half select

    v8f acc[4][2] = {};

    auto stage = [&](int buf, int kb) {
        // ---- A: 512 x 16B async DMA copies straight into LDS (ASYNCcnt) ----
#pragma unroll
        for (int i = 0; i < 2; ++i) {
            int v   = tid + i * 256;
            int off = v * 8;
            int r   = off >> 5;
            int c   = off & 31;
            async_copy_b128(lds_offset(&As[buf][off]),
                            &A[(size_t)(row0 + r) * K + kb + c]);
        }
        // ---- B: batch all 16 b32 loads, then convert+pack+store ----
        float b0[8], b1[8];
#pragma unroll
        for (int i = 0; i < 8; ++i) {
            int p  = tid + i * 256;     // 2048 (k-pair, n) pairs
            int k2 = p >> 7;
            int n  = p & 127;
            const float* src = &Bg[(size_t)(kb + k2 * 2) * N + col0 + n];
            b0[i] = src[0];
            b1[i] = src[N];
        }
#pragma unroll
        for (int i = 0; i < 8; ++i) {
            int p  = tid + i * 256;
            int k2 = p >> 7;
            int n  = p & 127;
            *(unsigned*)&Bs[buf][n * 32 + k2 * 2] = f2bf_pk(b0[i], b1[i]);
        }
        // prefetch the slab after next of the dominant B stream into L2
        if (kb + 64 < K) {
            int k = (tid >> 7) * 2;
            int n = tid & 127;
            __builtin_prefetch(&Bg[(size_t)(kb + 64 + k) * N + col0 + n], 0, 1);
        }
    };

    stage(0, 0);

    for (int kb = 0; kb < K; kb += 32) {
        const int cur = (kb >> 5) & 1;
        wait_asynccnt0();  // async A-tile copies for `cur` have landed in LDS
        __syncthreads();   // publish `cur`; prior compute done with cur^1
        if (kb + 32 < K) stage(cur ^ 1, kb + 32);

        // ---- per-lane fragment gathers (ISA 16-bit A / B VGPR layouts) ----
        V16 afr[4];
#pragma unroll
        for (int i = 0; i < 4; ++i) {
            int m = wm * 64 + i * 16 + mlo;
            v8u16 lo = *(const v8u16*)&As[cur][m * 32 + hi * 8];
            v8u16 h2 = *(const v8u16*)&As[cur][m * 32 + 16 + hi * 8];
            afr[i].u = __builtin_shufflevector(lo, h2, 0, 1, 2, 3, 4, 5, 6, 7,
                                               8, 9, 10, 11, 12, 13, 14, 15);
        }
        V16 bfr[2];
#pragma unroll
        for (int j = 0; j < 2; ++j) {
            int n = wn * 32 + j * 16 + mlo;
            v8u16 lo = *(const v8u16*)&Bs[cur][n * 32 + hi * 8];
            v8u16 h2 = *(const v8u16*)&Bs[cur][n * 32 + 16 + hi * 8];
            bfr[j].u = __builtin_shufflevector(lo, h2, 0, 1, 2, 3, 4, 5, 6, 7,
                                               8, 9, 10, 11, 12, 13, 14, 15);
        }

#pragma unroll
        for (int i = 0; i < 4; ++i)
#pragma unroll
            for (int j = 0; j < 2; ++j)
                acc[i][j] = __builtin_amdgcn_wmma_f32_16x16x32_bf16(
                    false, afr[i].b, false, bfr[j].b, (short)0, acc[i][j],
                    false, false);
    }

    // ---- store: C/D layout — VGPR v: lanes0-15 row v, lanes16-31 row v+8 ----
#pragma unroll
    for (int i = 0; i < 4; ++i) {
#pragma unroll
        for (int j = 0; j < 2; ++j) {
            int gr0 = row0 + wm * 64 + i * 16 + hi * 8;
            int gc  = col0 + wn * 32 + j * 16 + mlo;
#pragma unroll
            for (int v = 0; v < 8; ++v) {
                float val = acc[i][j][v];
                if (RELU) val = fmaxf(val, 0.0f);
                size_t o = ((size_t)e * Mrows + (gr0 + v)) * N + gc;
                if (OUTBF) ((unsigned short*)Dall)[o] = f2bf(val);
                else       ((float*)Dall)[o] = val;
            }
        }
    }
}

// ---------------------------------------------------------------------------
// 5) combine: out[s] = gates1[s] * eo[slot[s]]  (0 for dropped tokens)
// ---------------------------------------------------------------------------
__global__ void __launch_bounds__(256)
combine_kernel(const float* __restrict__ eo, const int* __restrict__ slot,
               const float* __restrict__ gates1, float* __restrict__ out) {
    const int s  = blockIdx.x;
    const int sl = slot[s];
    float* orow = out + (size_t)s * MOE_M;
    if (sl >= 0) {
        const float g = gates1[s];
        const float* er = eo + (size_t)sl * MOE_M;
        for (int m = threadIdx.x; m < MOE_M; m += 256) orow[m] = g * er[m];
    } else {
        for (int m = threadIdx.x; m < MOE_M; m += 256) orow[m] = 0.0f;
    }
}

// ---------------------------------------------------------------------------
// launcher
// ---------------------------------------------------------------------------
extern "C" void kernel_launch(void* const* d_in, const int* in_sizes, int n_in,
                              void* d_out, int out_size, void* d_ws, size_t ws_size,
                              hipStream_t stream) {
    const float* xs = (const float*)d_in[0];  // [S, M] (B*T flattened)
    const float* wg = (const float*)d_in[1];  // [E, M]
    const float* w1 = (const float*)d_in[2];  // [E, M, H]
    const float* w2 = (const float*)d_in[3];  // [E, H, M]

    // workspace carve-up (16B aligned)
    char* ws = (char*)d_ws;
    size_t off = 0;
    float* me = (float*)(ws + off);            off += 256;
    int* idx  = (int*)(ws + off);              off += (size_t)MOE_S * 4;
    float* g1 = (float*)(ws + off);            off += (size_t)MOE_S * 4;
    int* slot = (int*)(ws + off);              off += (size_t)MOE_S * 4;
    int* inv  = (int*)(ws + off);              off += (size_t)MOE_E * MOE_C * 4;
    unsigned short* disp = (unsigned short*)(ws + off);
    off += (size_t)MOE_E * MOE_C * MOE_M * 2;  // 16 MB bf16
    unsigned short* hbuf = (unsigned short*)(ws + off);
    off += (size_t)MOE_E * MOE_C * MOE_H * 2;  // 64 MB bf16
    float* eo = (float*)(ws + off);
    off += (size_t)MOE_E * MOE_C * MOE_M * 4;  // 32 MB f32

    float* out  = (float*)d_out;
    float* laux = out + (size_t)MOE_S * MOE_M;

    // 0) zero me accumulator
    init_me_kernel<<<1, 32, 0, stream>>>(me);

    // 1) gating
    gate_kernel<<<MOE_S, 256, 0, stream>>>(xs, wg, idx, g1, me);

    // 2) per-expert scan -> slots, inverse map, l_aux
    scan_kernel<<<1, 256, 0, stream>>>(idx, me, slot, inv, laux);

    // 3) dispatch (fp32 -> bf16 scatter)
    dispatch_kernel<<<MOE_E * MOE_C, 256, 0, stream>>>(xs, inv, disp);

    // 4) expert FFN: h = relu(disp @ w1) -> bf16, then eo = h @ w2 -> f32
    {
        dim3 grid1(MOE_H / 128, MOE_C / 128, MOE_E);  // (64, 4, 8)
        moe_gemm<true, true><<<grid1, 256, 0, stream>>>(disp, w1, (void*)hbuf,
                                                        MOE_C, MOE_H, MOE_M);
        dim3 grid2(MOE_M / 128, MOE_C / 128, MOE_E);  // (16, 4, 8)
        moe_gemm<false, false><<<grid2, 256, 0, stream>>>(hbuf, w2, (void*)eo,
                                                          MOE_C, MOE_M, MOE_H);
    }

    // 5) combine
    combine_kernel<<<MOE_S, 256, 0, stream>>>(eo, slot, g1, out);
}
